// MMDiTBlock_79671643341177
// MI455X (gfx1250) — matrix operand order repaired and checked by reading, not compiled, not run on hardware
//
#include <hip/hip_runtime.h>
#include <hip/hip_bf16.h>

// ---------------------------------------------------------------------------
// MMDiT block for MI455X (gfx1250, wave32, WMMA).
// All matmuls use v_wmma_f32_16x16x32_bf16 (bf16 in, f32 accumulate).
// Weights are pre-transposed to [N][K] bf16 once per call so GEMM B-fragments
// are two 16B runs per lane and load straight from the 192MB L2.
// ---------------------------------------------------------------------------

#define B_    4
#define NX_   1024
#define NC_   154
#define D_    1536
#define H_    24
#define HD_   64
#define MH_   6144
#define NT_   (NX_ + NC_)    // 1178
#define MODW_ (6 * D_)       // 9216

typedef __attribute__((ext_vector_type(16))) __bf16 v16bf;
typedef __attribute__((ext_vector_type(8)))  float  v8f;

static __device__ __forceinline__ __bf16 f2bf(float f) { return (__bf16)f; }

#define WMMA_BF16(a, b, c) \
  __builtin_amdgcn_wmma_f32_16x16x32_bf16(false, (a), false, (b), (short)0, (c), false, false)

// ----------------- tiled transpose + f32 -> bf16 (weights) -----------------
// in: [K][N] f32  ->  out: [N][K] bf16.  32x32 tiles through LDS, both sides
// coalesced.  K, N always multiples of 32 here.
__global__ __launch_bounds__(256) void transpose_cvt_kernel(
    const float* __restrict__ in, __bf16* __restrict__ out, int K, int N) {
  __shared__ __bf16 t[32][33];
  int n0 = blockIdx.x * 32, k0 = blockIdx.y * 32;
  int tx = threadIdx.x & 31, g = threadIdx.x >> 5;   // g = 0..7
#pragma unroll
  for (int i = 0; i < 4; ++i) {
    int k = g * 4 + i;
    t[k][tx] = f2bf(in[(size_t)(k0 + k) * N + n0 + tx]);
  }
  __syncthreads();
#pragma unroll
  for (int i = 0; i < 4; ++i) {
    int n = g * 4 + i;
    out[(size_t)(n0 + n) * K + k0 + tx] = t[tx][n];
  }
}

// ------------------------------- silu(vec) ---------------------------------
__global__ __launch_bounds__(256) void silu_kernel(const float* __restrict__ in,
                                                   float* __restrict__ out, int n) {
  int i = blockIdx.x * 256 + threadIdx.x;
  if (i < n) {
    float x = in[i];
    out[i] = x / (1.0f + __expf(-x));
  }
}

// -------------------- mod = silu(vec) @ w_mod + b_mod ----------------------
__global__ __launch_bounds__(256) void mod_gemm_kernel(const float* __restrict__ sv,
                                                       const float* __restrict__ wmod,
                                                       const float* __restrict__ bmod,
                                                       float* __restrict__ mod) {
  int t = blockIdx.x * 256 + threadIdx.x;   // 0 .. 36863
  int b = t / MODW_;
  int j = t - b * MODW_;
  const float* v = sv + b * D_;
  float acc = bmod[j];
  for (int d = 0; d < D_; ++d) acc = fmaf(v[d], wmod[(size_t)d * MODW_ + j], acc);
  mod[t] = acc;
}

// ---------------- LayerNorm (+ optional modulation) -> bf16 ----------------
__global__ __launch_bounds__(256) void ln_mod_kernel(const float* __restrict__ in,
                                                     __bf16* __restrict__ out,
                                                     const float* __restrict__ mod,
                                                     int sh_off, int sc_off,
                                                     int tokper, int has_mod) {
  int row = blockIdx.x, tid = threadIdx.x;
  const float* p = in + (size_t)row * D_;
  float vals[6], s = 0.f, s2 = 0.f;
#pragma unroll
  for (int i = 0; i < 6; ++i) {
    float v = p[tid + i * 256];
    vals[i] = v; s += v; s2 += v * v;
  }
  __shared__ float r1[256], r2[256];
  r1[tid] = s; r2[tid] = s2;
  __syncthreads();
  for (int st = 128; st > 0; st >>= 1) {
    if (tid < st) { r1[tid] += r1[tid + st]; r2[tid] += r2[tid + st]; }
    __syncthreads();
  }
  float mean = r1[0] * (1.0f / D_);
  float var  = r2[0] * (1.0f / D_) - mean * mean;
  float inv  = rsqrtf(var + 1e-6f);
  const float* mrow = mod + (size_t)(row / tokper) * MODW_;
#pragma unroll
  for (int i = 0; i < 6; ++i) {
    int col = tid + i * 256;
    float v = (vals[i] - mean) * inv;
    if (has_mod) v = v * (1.0f + mrow[sc_off + col]) + mrow[sh_off + col];
    out[(size_t)row * D_ + col] = f2bf(v);
  }
}

// --------------------------- tiled WMMA GEMM -------------------------------
// C = A[M,K](bf16) @ W (given as WT[N][K] bf16) + bias.
//   EPI 0: bf16 store           (QKV)
//   EPI 1: gelu(tanh) -> bf16   (MLP layer 1)
//   EPI 2: f32 = res + gate*(.) (attention proj, per-batch gate row)
//   EPI 3: f32 = res + (.)      (MLP layer 2, final residual)
// Block 256 thr = 8 waves (4m x 2n); tile 128x128, BK=64.
// A tile double-buffered in LDS; B fragments direct from global (L2-resident
// transposed weights, 2 x b128 per lane per fragment).
template <int EPI>
__global__ __launch_bounds__(256) void gemm_bf16_wmma(
    const __bf16* __restrict__ A, const __bf16* __restrict__ WT,
    const float* __restrict__ bias,
    float* __restrict__ outf, __bf16* __restrict__ outb,
    const float* __restrict__ res, const float* __restrict__ gate,
    int tokper, int M, int N, int K) {
  __shared__ __bf16 As[2][128][64];   // 2 x 16KB
  int tid = threadIdx.x, lane = tid & 31, wid = tid >> 5;
  int wm = wid & 3, wn = wid >> 2;
  int bm0 = blockIdx.y * 128, bn0 = blockIdx.x * 128;
  int c0 = (lane >> 4) * 8;     // A-frag K half offset
  int kbo = (lane >> 4) * 16;   // unused (B-frag K offset is c0)
  int nl = lane & 15;

  // cooperative A tile: 128 rows x 64 k = 16KB; 4 x b128 per thread
  auto loadA = [&](int k0, uint4 (&r)[4]) {
#pragma unroll
    for (int i = 0; i < 4; ++i) {
      int idx = tid + i * 256;                 // 0..1023
      int row = bm0 + (idx >> 3);
      if (row >= M) row = M - 1;
      r[i] = *(const uint4*)&A[(size_t)row * K + k0 + ((idx & 7) << 3)];
    }
  };
  auto storeA = [&](int buf, const uint4 (&r)[4]) {
#pragma unroll
    for (int i = 0; i < 4; ++i) {
      int idx = tid + i * 256;
      *(uint4*)&As[buf][idx >> 3][(idx & 7) << 3] = r[i];
    }
  };

  v8f acc[2][4] = {};
  uint4 rg[4];
  loadA(0, rg);
  storeA(0, rg);
  __syncthreads();

  const int T = K >> 6;
  for (int t = 0; t < T; ++t) {
    int k0 = t << 6;
    uint4 nxt[4];
    if (t + 1 < T) {
      loadA(k0 + 64, nxt);
      // prefetch next weight tile into L2/WGP$ (global_prefetch_b8)
      __builtin_prefetch(WT + (size_t)(bn0 + wn * 64 + nl) * K + k0 + 64, 0, 0);
    }
#pragma unroll
    for (int ks = 0; ks < 2; ++ks) {
      v16bf af[2];
#pragma unroll
      for (int i = 0; i < 2; ++i) {
        int m = wm * 32 + i * 16 + nl;
        ((uint4*)&af[i])[0] = *(const uint4*)&As[t & 1][m][ks * 32 + c0];
        ((uint4*)&af[i])[1] = *(const uint4*)&As[t & 1][m][ks * 32 + c0 + 16];
      }
#pragma unroll
      for (int j = 0; j < 4; ++j) {
        const __bf16* wp =
            WT + (size_t)(bn0 + wn * 64 + j * 16 + nl) * K + k0 + ks * 32 + c0;
        v16bf bv;
        ((uint4*)&bv)[0] = *(const uint4*)(wp);
        ((uint4*)&bv)[1] = *(const uint4*)(wp + 16);
#pragma unroll
        for (int i = 0; i < 2; ++i)
          acc[i][j] = WMMA_BF16(af[i], bv, acc[i][j]);
      }
    }
    if (t + 1 < T) storeA((t + 1) & 1, nxt);
    __syncthreads();
  }

  // Epilogue. C layout: VGPR r -> row r (lanes 0-15) / r+8 (lanes 16-31),
  // col = lane&15 within each 16x16 tile.
  int rlo = (lane >> 4) * 8;
#pragma unroll
  for (int i = 0; i < 2; ++i) {
#pragma unroll
    for (int j = 0; j < 4; ++j) {
      int col = bn0 + wn * 64 + j * 16 + nl;
      float bv = bias[col];
#pragma unroll
      for (int r = 0; r < 8; ++r) {
        int row = bm0 + wm * 32 + i * 16 + rlo + r;
        if (row >= M) continue;
        float v = acc[i][j][r] + bv;
        size_t off = (size_t)row * N + col;
        if (EPI == 0) {
          outb[off] = f2bf(v);
        } else if (EPI == 1) {
          float u = 0.7978845608028654f * (v + 0.044715f * v * v * v);
          outb[off] = f2bf(0.5f * v * (1.0f + tanhf(u)));
        } else if (EPI == 2) {
          float g = gate[(size_t)(row / tokper) * MODW_ + col];
          outf[off] = res[off] + g * v;
        } else {
          outf[off] = res[off] + v;
        }
      }
    }
  }
}

// ----------------------------- flash attention -----------------------------
// One wave per (b, h, 16-query tile); keys streamed in tiles of 32.
// QK^T: 4 WMMA, P*V: 4 WMMA per tile. V transposed through LDS; P converted
// C-layout -> A-layout through LDS.
__global__ __launch_bounds__(32) void attn_kernel(
    const __bf16* __restrict__ qkv_x, const __bf16* __restrict__ qkv_c,
    __bf16* __restrict__ o_x, __bf16* __restrict__ o_c) {
  const int QT = (NT_ + 15) / 16;   // 74
  int lane = threadIdx.x;
  int blk = blockIdx.x;
  int qt = blk % QT, bh = blk / QT;
  int h = bh % H_, b = bh / H_;
  int nl = lane & 15, c0 = (lane >> 4) * 8, kbo = (lane >> 4) * 16;

  auto tok = [&](int s, int comp) -> const __bf16* {
    if (s >= NT_) s = NT_ - 1;
    if (s < NX_)
      return qkv_x + ((size_t)((b * NX_ + s) * 3 + comp)) * D_ + h * HD_;
    return qkv_c + ((size_t)((b * NC_ + (s - NX_)) * 3 + comp)) * D_ + h * HD_;
  };

  const __bf16* qp = tok(qt * 16 + nl, 0);
  v16bf qf[2];
#pragma unroll
  for (int i = 0; i < 2; ++i) {
    ((uint4*)&qf[i])[0] = *(const uint4*)(qp + 32 * i + c0);
    ((uint4*)&qf[i])[1] = *(const uint4*)(qp + 32 * i + c0 + 16);
  }

  float mrun[8], lrun[8];
#pragma unroll
  for (int r = 0; r < 8; ++r) { mrun[r] = -1e30f; lrun[r] = 0.f; }
  v8f oacc[4] = {};

  __shared__ __bf16 P[16][32];
  __shared__ __bf16 Vs[64][32];   // [hd][key] transposed
  const float scale = 0.125f;     // 64^-0.5
  const int KT = (NT_ + 31) / 32; // 37

  for (int kt = 0; kt < KT; ++kt) {
    v8f sc2[2] = {};
#pragma unroll
    for (int hn = 0; hn < 2; ++hn) {
      const __bf16* kp = tok(kt * 32 + hn * 16 + nl, 1);
#pragma unroll
      for (int i = 0; i < 2; ++i) {
        v16bf kf;   // B layout: lane col = key, hd c0..c0+7 and c0+16..c0+23 per lane
        ((uint4*)&kf)[0] = *(const uint4*)(kp + 32 * i + c0);
        ((uint4*)&kf)[1] = *(const uint4*)(kp + 32 * i + c0 + 16);
        sc2[hn] = WMMA_BF16(qf[i], kf, sc2[hn]);
      }
    }

    // stage V tile (32 keys x 64 hd) transposed into LDS
#pragma unroll
    for (int it = 0; it < 8; ++it) {
      int idx = lane + it * 32;          // 0..255
      int key = idx >> 3, hd0 = (idx & 7) << 3;
      const __bf16* vp = tok(kt * 32 + key, 2);
      union { uint4 u; __bf16 hh[8]; } t;
      t.u = *(const uint4*)(vp + hd0);
#pragma unroll
      for (int e = 0; e < 8; ++e) Vs[hd0 + e][key] = t.hh[e];
    }

    // online softmax (16-lane row reductions)
    int kbase = kt * 32;
#pragma unroll
    for (int r = 0; r < 8; ++r) {
      float s0 = sc2[0][r] * scale, s1 = sc2[1][r] * scale;
      if (kbase + nl >= NT_)      s0 = -1e30f;
      if (kbase + 16 + nl >= NT_) s1 = -1e30f;
      float mx = fmaxf(s0, s1);
      for (int t = 1; t < 16; t <<= 1) mx = fmaxf(mx, __shfl_xor(mx, t, 16));
      float mnew = fmaxf(mrun[r], mx);
      float p0 = __expf(s0 - mnew), p1 = __expf(s1 - mnew);
      float rs = p0 + p1;
      for (int t = 1; t < 16; t <<= 1) rs += __shfl_xor(rs, t, 16);
      float alpha = __expf(mrun[r] - mnew);
      lrun[r] = lrun[r] * alpha + rs;
      mrun[r] = mnew;
#pragma unroll
      for (int t2 = 0; t2 < 4; ++t2) oacc[t2][r] *= alpha;
      int prow = r + ((lane >> 4) << 3);
      P[prow][nl]      = f2bf(p0);
      P[prow][16 + nl] = f2bf(p1);
    }
    __syncthreads();

    v16bf pf;
    ((uint4*)&pf)[0] = *(const uint4*)&P[nl][c0];
    ((uint4*)&pf)[1] = *(const uint4*)&P[nl][c0 + 16];
#pragma unroll
    for (int t2 = 0; t2 < 4; ++t2) {
      v16bf vf;
      int hd = t2 * 16 + nl;
      ((uint4*)&vf)[0] = *(const uint4*)&Vs[hd][c0];
      ((uint4*)&vf)[1] = *(const uint4*)&Vs[hd][c0 + 16];
      oacc[t2] = WMMA_BF16(pf, vf, oacc[t2]);
    }
    __syncthreads();
  }

#pragma unroll
  for (int t2 = 0; t2 < 4; ++t2) {
    int hd = t2 * 16 + nl;
#pragma unroll
    for (int r = 0; r < 8; ++r) {
      int row = r + ((lane >> 4) << 3);
      int s = qt * 16 + row;
      if (s >= NT_) continue;
      float v = oacc[t2][r] / lrun[r];
      if (s < NX_)
        o_x[((size_t)(b * NX_ + s)) * D_ + h * HD_ + hd] = f2bf(v);
      else
        o_c[((size_t)(b * NC_ + (s - NX_))) * D_ + h * HD_ + hd] = f2bf(v);
    }
  }
}

// ---------------------------------------------------------------------------
extern "C" void kernel_launch(void* const* d_in, const int* in_sizes, int n_in,
                              void* d_out, int out_size, void* d_ws, size_t ws_size,
                              hipStream_t stream) {
  const float* x      = (const float*)d_in[0];
  const float* c      = (const float*)d_in[1];
  const float* vec    = (const float*)d_in[2];
  const float* w_mod  = (const float*)d_in[3];
  const float* b_mod  = (const float*)d_in[4];
  const float* w_qkvx = (const float*)d_in[5];
  const float* b_qkvx = (const float*)d_in[6];
  const float* w_qkvc = (const float*)d_in[7];
  const float* b_qkvc = (const float*)d_in[8];
  const float* w_px   = (const float*)d_in[9];
  const float* b_px   = (const float*)d_in[10];
  const float* w_pc   = (const float*)d_in[11];
  const float* b_pc   = (const float*)d_in[12];
  const float* w1x    = (const float*)d_in[13];
  const float* b1x    = (const float*)d_in[14];
  const float* w2x    = (const float*)d_in[15];
  const float* b2x    = (const float*)d_in[16];
  const float* w1c    = (const float*)d_in[17];
  const float* b1c    = (const float*)d_in[18];
  const float* w2c    = (const float*)d_in[19];
  const float* b2c    = (const float*)d_in[20];

  const int MX = B_ * NX_;   // 4096
  const int MC = B_ * NC_;   // 616

  size_t off = 0;
  auto alloc = [&](size_t bytes) {
    void* p = (char*)d_ws + off;
    off = (off + bytes + 255) & ~(size_t)255;
    return p;
  };
  float*  silu_v = (float*)alloc((size_t)B_ * D_ * 4);
  float*  modb   = (float*)alloc((size_t)B_ * MODW_ * 4);
  __bf16* wqx_t  = (__bf16*)alloc((size_t)D_ * 3 * D_ * 2);   // [3D][D]
  __bf16* wqc_t  = (__bf16*)alloc((size_t)D_ * 3 * D_ * 2);
  __bf16* wpx_t  = (__bf16*)alloc((size_t)D_ * D_ * 2);
  __bf16* wpc_t  = (__bf16*)alloc((size_t)D_ * D_ * 2);
  __bf16* w1x_t  = (__bf16*)alloc((size_t)D_ * MH_ * 2);      // [MH][D]
  __bf16* w1c_t  = (__bf16*)alloc((size_t)D_ * MH_ * 2);
  __bf16* w2x_t  = (__bf16*)alloc((size_t)MH_ * D_ * 2);      // [D][MH]
  __bf16* w2c_t  = (__bf16*)alloc((size_t)MH_ * D_ * 2);
  __bf16* xn     = (__bf16*)alloc((size_t)MX * D_ * 2);
  __bf16* cn     = (__bf16*)alloc((size_t)MC * D_ * 2);
  __bf16* qkvx   = (__bf16*)alloc((size_t)MX * 3 * D_ * 2);
  __bf16* qkvc   = (__bf16*)alloc((size_t)MC * 3 * D_ * 2);
  __bf16* ox     = (__bf16*)alloc((size_t)MX * D_ * 2);
  __bf16* oc     = (__bf16*)alloc((size_t)MC * D_ * 2);
  float*  x1     = (float*)alloc((size_t)MX * D_ * 4);
  float*  c1     = (float*)alloc((size_t)MC * D_ * 4);
  __bf16* ln1    = (__bf16*)alloc((size_t)MX * D_ * 2);
  __bf16* hbuf   = (__bf16*)alloc((size_t)MX * MH_ * 2);
  (void)ws_size; (void)n_in; (void)in_sizes; (void)out_size;

  // transpose + convert weights: in [K][N] f32 -> out [N][K] bf16
  auto cvtT = [&](const float* src, __bf16* dst, int K, int N) {
    transpose_cvt_kernel<<<dim3(N / 32, K / 32), 256, 0, stream>>>(src, dst, K, N);
  };
  cvtT(w_qkvx, wqx_t, D_, 3 * D_);
  cvtT(w_qkvc, wqc_t, D_, 3 * D_);
  cvtT(w_px,   wpx_t, D_, D_);
  cvtT(w_pc,   wpc_t, D_, D_);
  cvtT(w1x,    w1x_t, D_, MH_);
  cvtT(w1c,    w1c_t, D_, MH_);
  cvtT(w2x,    w2x_t, MH_, D_);
  cvtT(w2c,    w2c_t, MH_, D_);

  silu_kernel<<<(B_ * D_ + 255) / 256, 256, 0, stream>>>(vec, silu_v, B_ * D_);
  mod_gemm_kernel<<<(B_ * MODW_) / 256, 256, 0, stream>>>(silu_v, w_mod, b_mod, modb);

  // LN + modulation -> bf16 activations
  ln_mod_kernel<<<MX, 256, 0, stream>>>(x, xn, modb, 0,      D_,     NX_, 1);
  ln_mod_kernel<<<MC, 256, 0, stream>>>(c, cn, modb, 3 * D_, 4 * D_, NC_, 1);

  // QKV projections
  gemm_bf16_wmma<0><<<dim3(3 * D_ / 128, (MX + 127) / 128), 256, 0, stream>>>(
      xn, wqx_t, b_qkvx, nullptr, qkvx, nullptr, nullptr, NX_, MX, 3 * D_, D_);
  gemm_bf16_wmma<0><<<dim3(3 * D_ / 128, (MC + 127) / 128), 256, 0, stream>>>(
      cn, wqc_t, b_qkvc, nullptr, qkvc, nullptr, nullptr, NC_, MC, 3 * D_, D_);

  // joint attention over NX+NC tokens
  attn_kernel<<<B_ * H_ * ((NT_ + 15) / 16), 32, 0, stream>>>(qkvx, qkvc, ox, oc);

  // output projections with gated residual: x1 = x + g_x * (o @ w_proj + b)
  gemm_bf16_wmma<2><<<dim3(D_ / 128, (MX + 127) / 128), 256, 0, stream>>>(
      ox, wpx_t, b_px, x1, nullptr, x, modb + 2 * D_, NX_, MX, D_, D_);
  gemm_bf16_wmma<2><<<dim3(D_ / 128, (MC + 127) / 128), 256, 0, stream>>>(
      oc, wpc_t, b_pc, c1, nullptr, c, modb + 5 * D_, NC_, MC, D_, D_);

  float* outx = (float*)d_out;
  float* outc = (float*)d_out + (size_t)MX * D_;

  // MLP (x): x_out = x1 + gelu(ln(x1) @ w1 + b1) @ w2 + b2
  ln_mod_kernel<<<MX, 256, 0, stream>>>(x1, ln1, modb, 0, 0, NX_, 0);
  gemm_bf16_wmma<1><<<dim3(MH_ / 128, (MX + 127) / 128), 256, 0, stream>>>(
      ln1, w1x_t, b1x, nullptr, hbuf, nullptr, nullptr, NX_, MX, MH_, D_);
  gemm_bf16_wmma<3><<<dim3(D_ / 128, (MX + 127) / 128), 256, 0, stream>>>(
      hbuf, w2x_t, b2x, outx, nullptr, x1, nullptr, NX_, MX, D_, MH_);

  // MLP (c)
  ln_mod_kernel<<<MC, 256, 0, stream>>>(c1, ln1, modb, 0, 0, NC_, 0);
  gemm_bf16_wmma<1><<<dim3(MH_ / 128, (MC + 127) / 128), 256, 0, stream>>>(
      ln1, w1c_t, b1c, nullptr, hbuf, nullptr, nullptr, NC_, MC, MH_, D_);
  gemm_bf16_wmma<3><<<dim3(D_ / 128, (MC + 127) / 128), 256, 0, stream>>>(
      hbuf, w2c_t, b2c, outc, nullptr, c1, nullptr, NC_, MC, D_, MH_);
}
